// GlobalLinearSelfAttention_54606214202167
// MI455X (gfx1250) — matrix operand order repaired and checked by reading, not compiled, not run on hardware
//
#include <hip/hip_runtime.h>

// ---------------------------------------------------------------------------
// GlobalLinearSelfAttention for MI455X (gfx1250, wave32, WMMA).
// All matmuls use v_wmma_f32_16x16x32_bf16 (bf16 in, fp32 accumulate).
// B=4, N=4096, DIM=1024, HEADS=16, DIM_HEAD=64, INNER=1024, SCALE=1/8.
// Tile staging uses GLOBAL_LOAD_ASYNC_TO_LDS_B128 (ASYNCcnt-tracked,
// VGPR-bypassing) -- the builtin takes int4 pointers in AS1/AS3.
// ---------------------------------------------------------------------------

typedef __bf16 bf16;
typedef __attribute__((ext_vector_type(16))) __bf16 v16bf;
typedef __attribute__((ext_vector_type(8)))  __bf16 v8bf;
typedef __attribute__((ext_vector_type(4)))  __bf16 v4bf;
typedef __attribute__((ext_vector_type(8)))  float  v8f;
typedef __attribute__((ext_vector_type(4)))  float  v4f;
typedef __attribute__((ext_vector_type(4)))  int    v4i;

typedef __attribute__((address_space(1))) v4i g_v4i;   // global
typedef __attribute__((address_space(3))) v4i l_v4i;   // LDS

#define DEV __device__ __forceinline__

#if defined(__has_builtin)
#  if __has_builtin(__builtin_amdgcn_global_load_async_to_lds_b128)
#    define HAVE_ASYNC_LDS 1
#  endif
#endif
#ifndef HAVE_ASYNC_LDS
#  define HAVE_ASYNC_LDS 0
#endif

union ABfrag { v16bf v; v8bf h[2]; };

// ---- global -> LDS staging (16B per call, per-lane addresses) -------------
DEV void stage16(const bf16* g, bf16* l) {
#if HAVE_ASYNC_LDS
  __builtin_amdgcn_global_load_async_to_lds_b128(
      (g_v4i*)g, (l_v4i*)l, 0, 0);
#else
  *(v8bf*)l = *(const v8bf*)g;
#endif
}
DEV void stage_commit() {
#if HAVE_ASYNC_LDS
#  if __has_builtin(__builtin_amdgcn_s_wait_asynccnt)
  __builtin_amdgcn_s_wait_asynccnt(0);
#  else
  asm volatile("s_wait_asynccnt 0x0" ::: "memory");
#  endif
#endif
}

// ---- wave32 reductions via ds_swizzle (xor-mask group-of-32 patterns) -----
DEV float wave_max(float x) {
  x = fmaxf(x, __int_as_float(__builtin_amdgcn_ds_swizzle(__float_as_int(x), 0x401F))); // xor 16
  x = fmaxf(x, __int_as_float(__builtin_amdgcn_ds_swizzle(__float_as_int(x), 0x201F))); // xor 8
  x = fmaxf(x, __int_as_float(__builtin_amdgcn_ds_swizzle(__float_as_int(x), 0x101F))); // xor 4
  x = fmaxf(x, __int_as_float(__builtin_amdgcn_ds_swizzle(__float_as_int(x), 0x081F))); // xor 2
  x = fmaxf(x, __int_as_float(__builtin_amdgcn_ds_swizzle(__float_as_int(x), 0x041F))); // xor 1
  return x;
}
DEV float wave_sum(float x) {
  x += __int_as_float(__builtin_amdgcn_ds_swizzle(__float_as_int(x), 0x401F));
  x += __int_as_float(__builtin_amdgcn_ds_swizzle(__float_as_int(x), 0x201F));
  x += __int_as_float(__builtin_amdgcn_ds_swizzle(__float_as_int(x), 0x101F));
  x += __int_as_float(__builtin_amdgcn_ds_swizzle(__float_as_int(x), 0x081F));
  x += __int_as_float(__builtin_amdgcn_ds_swizzle(__float_as_int(x), 0x041F));
  return x;
}

// ---- WMMA fragment loaders (ISA 7.12.2 layouts) ---------------------------
// A: 16x32 bf16 row-major, stride elems. lane<16: row=lane, K 0..7 & 16..23;
// lane>=16: row=lane-16, K 8..15 & 24..31.  Two b128 loads per lane.
DEV v16bf load_frag_a(const bf16* __restrict__ base, int stride, int lane) {
  const int l  = lane & 15;
  const int ko = (lane >> 4) * 8;
  const bf16* r = base + (size_t)l * stride;
  ABfrag f;
  f.h[0] = *(const v8bf*)(r + ko);
  f.h[1] = *(const v8bf*)(r + 16 + ko);
  return f.v;
}
// B stored as B^T row-major [n][k] (stride elems). B is 32x16:
// lane<16: col=lane, K 0..7 & 16..23; lane>=16: col=lane-16, K 8..15 & 24..31.
DEV v16bf load_frag_b(const bf16* __restrict__ base, int stride, int lane) {
  const int l = lane & 15;
  const bf16* r = base + (size_t)l * stride + (lane >> 4) * 8;
  ABfrag f;
  f.h[0] = *(const v8bf*)(r);
  f.h[1] = *(const v8bf*)(r + 16);
  return f.v;
}

DEV v8f wmma_bf16(v16bf a, v16bf b, v8f c) {
  return __builtin_amdgcn_wmma_f32_16x16x32_bf16(false, a, false, b, (short)0, c,
                                                 false, false);
}

// ---------------------------------------------------------------------------
// K0: fp32 -> bf16 convert (n divisible by 1024; 4 elems/thread)
// ---------------------------------------------------------------------------
__global__ __launch_bounds__(256) void k_cvt(const float* __restrict__ src,
                                             bf16* __restrict__ dst) {
  const size_t i = ((size_t)blockIdx.x * 256 + threadIdx.x) * 4;
  v4f x = *(const v4f*)(src + i);
  v4bf y;
  y[0] = (bf16)x[0]; y[1] = (bf16)x[1]; y[2] = (bf16)x[2]; y[3] = (bf16)x[3];
  *(v4bf*)(dst + i) = y;
}

// ---------------------------------------------------------------------------
// K1: QKV GEMM.  C[16384,3072] = feats_bf[16384,1024] @ wqkv_bf[3072,1024]^T
// 128x128 tile per block (256 thr / 8 waves, each wave 32x64 = 2x4 tiles).
// Epilogue scatters: q -> fp32 [b,h,n,64];  k -> fp32 [b,h,d,n] (masked -inf);
// v -> bf16 [b,h,e,n] (masked 0).
// ---------------------------------------------------------------------------
__global__ __launch_bounds__(256) void k_gemm_qkv(
    const bf16* __restrict__ A, const bf16* __restrict__ W,
    const unsigned char* __restrict__ mask,
    float* __restrict__ qf, float* __restrict__ kt, bf16* __restrict__ vt) {
  __shared__ bf16 As[128 * 32];
  __shared__ bf16 Bs[128 * 32];
  const int tid  = threadIdx.x;
  const int lane = tid & 31, wave = tid >> 5;
  const int wm = wave & 3, wn = wave >> 2;      // 4x2 wave grid
  const int mBase = blockIdx.x * 128;
  const int nBase = blockIdx.y * 128;
  const int sr = tid >> 2, ss = (tid & 3) * 8;  // staging row/segment

  v8f acc[2][4] = {};
  for (int k0 = 0; k0 < 1024; k0 += 32) {
    // straight-line staging: 4 x 16B per thread (rows sr and sr+64 of each tile)
    stage16(A + (size_t)(mBase + sr)      * 1024 + k0 + ss, As + sr * 32 + ss);
    stage16(A + (size_t)(mBase + sr + 64) * 1024 + k0 + ss, As + (sr + 64) * 32 + ss);
    stage16(W + (size_t)(nBase + sr)      * 1024 + k0 + ss, Bs + sr * 32 + ss);
    stage16(W + (size_t)(nBase + sr + 64) * 1024 + k0 + ss, Bs + (sr + 64) * 32 + ss);
    stage_commit();
    __syncthreads();
    v16bf af[2], bfr[4];
#pragma unroll
    for (int i = 0; i < 2; ++i) af[i] = load_frag_a(As + (wm * 32 + i * 16) * 32, 32, lane);
#pragma unroll
    for (int j = 0; j < 4; ++j) bfr[j] = load_frag_b(Bs + (wn * 64 + j * 16) * 32, 32, lane);
#pragma unroll
    for (int i = 0; i < 2; ++i)
#pragma unroll
      for (int j = 0; j < 4; ++j) acc[i][j] = wmma_bf16(af[i], bfr[j], acc[i][j]);
    __syncthreads();
  }

  const int hi = lane >> 4, col = lane & 15;
#pragma unroll
  for (int i = 0; i < 2; ++i)
#pragma unroll
    for (int j = 0; j < 4; ++j) {
      const int gcol  = nBase + wn * 64 + j * 16 + col;
      const int which = gcol >> 10;            // 0=q 1=k 2=v
      const int h     = (gcol >> 6) & 15;
      const int d     = gcol & 63;
      const int rbase = mBase + wm * 32 + i * 16 + hi * 8;
#pragma unroll
      for (int r = 0; r < 8; ++r) {
        const int grow = rbase + r;            // global row = b*4096+n
        const int b = grow >> 12, n = grow & 4095;
        const int bh = b * 16 + h;
        const float val = acc[i][j][r];
        if (which == 0) {
          qf[((size_t)bh * 4096 + n) * 64 + d] = val;
        } else {
          const bool mk = mask[grow] != 0;
          const size_t idx = ((size_t)bh * 64 + d) * 4096 + n;
          if (which == 1) kt[idx] = mk ? val : -3.402823466e38f;
          else            vt[idx] = (bf16)(mk ? val : 0.0f);
        }
      }
    }
}

// ---------------------------------------------------------------------------
// K2: k softmax over N (rows of kt): E[d,n] = exp(k - rowmax) (bf16), Z[d]=sum
// one wave per 4096-elem row; 8 rows/block; 512 blocks.
// ---------------------------------------------------------------------------
__global__ __launch_bounds__(256) void k_ksoft(const float* __restrict__ kt,
                                               bf16* __restrict__ Et,
                                               float* __restrict__ Z) {
  const int lane = threadIdx.x & 31, wave = threadIdx.x >> 5;
  const int row = blockIdx.x * 8 + wave;                  // bh*64 + d
  const float* src = kt + (size_t)row * 4096;
  float m = -3.402823466e38f;
  for (int it = 0; it < 32; ++it) {
    v4f x = *(const v4f*)(src + it * 128 + lane * 4);
    m = fmaxf(m, fmaxf(fmaxf(x[0], x[1]), fmaxf(x[2], x[3])));
  }
  m = wave_max(m);
  bf16* dst = Et + (size_t)row * 4096;
  float s = 0.f;
  for (int it = 0; it < 32; ++it) {
    const int o = it * 128 + lane * 4;
    v4f x = *(const v4f*)(src + o);
    const float e0 = __expf(x[0] - m), e1 = __expf(x[1] - m);
    const float e2 = __expf(x[2] - m), e3 = __expf(x[3] - m);
    s += (e0 + e1) + (e2 + e3);
    v4bf y; y[0] = (bf16)e0; y[1] = (bf16)e1; y[2] = (bf16)e2; y[3] = (bf16)e3;
    *(v4bf*)(dst + o) = y;
  }
  s = wave_sum(s);
  if (lane == 0) Z[row] = s;
}

// ---------------------------------------------------------------------------
// K2b: q softmax over dim_head (64) * SCALE -> bf16.  One wave per row.
// ---------------------------------------------------------------------------
__global__ __launch_bounds__(256) void k_qsoft(const float* __restrict__ qf,
                                               bf16* __restrict__ qb) {
  const int lane = threadIdx.x & 31, wave = threadIdx.x >> 5;
  const size_t row = (size_t)blockIdx.x * 8 + wave;       // 0..262143
  const float* src = qf + row * 64;
  const float a = src[lane], b = src[lane + 32];
  const float m = wave_max(fmaxf(a, b));
  const float e0 = __expf(a - m), e1 = __expf(b - m);
  const float s = wave_sum(e0 + e1);
  const float inv = 0.125f / s;                           // SCALE = 64^-0.5
  bf16* dst = qb + row * 64;
  dst[lane]      = (bf16)(e0 * inv);
  dst[lane + 32] = (bf16)(e1 * inv);
}

// ---------------------------------------------------------------------------
// K3: context per (b,h): U[d,e] = sum_n E[d,n]*V^T[e,n]; ctx = U / Z[d] (bf16)
// 64 blocks x 4 waves; wave owns one 16-row d-strip x 64 cols; K=4096.
// ---------------------------------------------------------------------------
__global__ __launch_bounds__(128) void k_ctx(const bf16* __restrict__ Et,
                                             const bf16* __restrict__ vt,
                                             const float* __restrict__ Z,
                                             bf16* __restrict__ ctx) {
  const int lane = threadIdx.x & 31, wave = threadIdx.x >> 5;
  const int bh = blockIdx.x;
  const bf16* Eb = Et + (size_t)bh * 64 * 4096;
  const bf16* Vb = vt + (size_t)bh * 64 * 4096;
  const int d0 = wave * 16;
  v8f acc[4] = {};
  for (int k0 = 0; k0 < 4096; k0 += 32) {
    v16bf af = load_frag_a(Eb + (size_t)d0 * 4096 + k0, 4096, lane);
#pragma unroll
    for (int j = 0; j < 4; ++j) {
      v16bf bfr = load_frag_b(Vb + (size_t)(j * 16) * 4096 + k0, 4096, lane);
      acc[j] = wmma_bf16(af, bfr, acc[j]);
    }
  }
  const int hi = lane >> 4, col = lane & 15;
#pragma unroll
  for (int j = 0; j < 4; ++j)
#pragma unroll
    for (int r = 0; r < 8; ++r) {
      const int d = d0 + r + hi * 8;
      const float val = acc[j][r] / Z[bh * 64 + d];
      ctx[((size_t)bh * 64 + d) * 64 + j * 16 + col] = (bf16)val;
    }
}

// ---------------------------------------------------------------------------
// K4: out[n,e] = sum_d q[n,d] * ctx[e,d]  (ctx row-major IS B^T already).
// Writes attn bf16 in [b, n, h*64+e] layout for the output projection.
// grid = 64 bh * 32 blocks; 8 waves x 16 rows; K=64.
// ---------------------------------------------------------------------------
__global__ __launch_bounds__(256) void k_attn(const bf16* __restrict__ qb,
                                              const bf16* __restrict__ ctx,
                                              bf16* __restrict__ attn) {
  const int lane = threadIdx.x & 31, wave = threadIdx.x >> 5;
  const int bh = blockIdx.x >> 5;
  const int n0 = ((blockIdx.x & 31) * 8 + wave) * 16;
  const int b = bh >> 4, h = bh & 15;
  const bf16* Ab = qb + ((size_t)bh * 4096 + n0) * 64;
  const bf16* Bb = ctx + (size_t)bh * 64 * 64;
  v8f acc[4] = {};
#pragma unroll
  for (int k0 = 0; k0 < 64; k0 += 32) {
    v16bf af = load_frag_a(Ab + k0, 64, lane);
#pragma unroll
    for (int j = 0; j < 4; ++j) {
      v16bf bfr = load_frag_b(Bb + (j * 16) * 64 + k0, 64, lane);
      acc[j] = wmma_bf16(af, bfr, acc[j]);
    }
  }
  const int hi = lane >> 4, col = lane & 15;
#pragma unroll
  for (int j = 0; j < 4; ++j)
#pragma unroll
    for (int r = 0; r < 8; ++r) {
      const int n = n0 + r + hi * 8;
      const int e = j * 16 + col;
      attn[((size_t)b * 4096 + n) * 1024 + h * 64 + e] = (bf16)acc[j][r];
    }
}

// ---------------------------------------------------------------------------
// K5: output projection: out[16384,1024] = attn @ w_out^T + b_out (fp32 out)
// ---------------------------------------------------------------------------
__global__ __launch_bounds__(256) void k_gemm_out(const bf16* __restrict__ A,
                                                  const bf16* __restrict__ W,
                                                  const float* __restrict__ bias,
                                                  float* __restrict__ out) {
  __shared__ bf16 As[128 * 32];
  __shared__ bf16 Bs[128 * 32];
  const int tid  = threadIdx.x;
  const int lane = tid & 31, wave = tid >> 5;
  const int wm = wave & 3, wn = wave >> 2;
  const int mBase = blockIdx.x * 128;
  const int nBase = blockIdx.y * 128;
  const int sr = tid >> 2, ss = (tid & 3) * 8;

  v8f acc[2][4] = {};
  for (int k0 = 0; k0 < 1024; k0 += 32) {
    stage16(A + (size_t)(mBase + sr)      * 1024 + k0 + ss, As + sr * 32 + ss);
    stage16(A + (size_t)(mBase + sr + 64) * 1024 + k0 + ss, As + (sr + 64) * 32 + ss);
    stage16(W + (size_t)(nBase + sr)      * 1024 + k0 + ss, Bs + sr * 32 + ss);
    stage16(W + (size_t)(nBase + sr + 64) * 1024 + k0 + ss, Bs + (sr + 64) * 32 + ss);
    stage_commit();
    __syncthreads();
    v16bf af[2], bfr[4];
#pragma unroll
    for (int i = 0; i < 2; ++i) af[i] = load_frag_a(As + (wm * 32 + i * 16) * 32, 32, lane);
#pragma unroll
    for (int j = 0; j < 4; ++j) bfr[j] = load_frag_b(Bs + (wn * 64 + j * 16) * 32, 32, lane);
#pragma unroll
    for (int i = 0; i < 2; ++i)
#pragma unroll
      for (int j = 0; j < 4; ++j) acc[i][j] = wmma_bf16(af[i], bfr[j], acc[i][j]);
    __syncthreads();
  }

  const int hi = lane >> 4, col = lane & 15;
#pragma unroll
  for (int i = 0; i < 2; ++i)
#pragma unroll
    for (int j = 0; j < 4; ++j) {
      const int gcol  = nBase + wn * 64 + j * 16 + col;
      const float bv  = bias[gcol];
      const int rbase = mBase + wm * 32 + i * 16 + hi * 8;
#pragma unroll
      for (int r = 0; r < 8; ++r)
        out[(size_t)(rbase + r) * 1024 + gcol] = acc[i][j][r] + bv;
    }
}

// ---------------------------------------------------------------------------
extern "C" void kernel_launch(void* const* d_in, const int* in_sizes, int n_in,
                              void* d_out, int out_size, void* d_ws, size_t ws_size,
                              hipStream_t stream) {
  (void)in_sizes; (void)n_in; (void)out_size; (void)ws_size;
  const float*         feats = (const float*)d_in[0];          // [4,4096,1024]
  const unsigned char* mask  = (const unsigned char*)d_in[1];  // [4,4096] bool
  const float*         w_qkv = (const float*)d_in[2];          // [3072,1024]
  const float*         w_out = (const float*)d_in[3];          // [1024,1024]
  const float*         b_out = (const float*)d_in[4];          // [1024]
  float*               out   = (float*)d_out;                  // [4,4096,1024]

  char* p = (char*)d_ws;
  bf16*  feats_bf = (bf16*)p;  p += (size_t)16384 * 1024 * 2;  // 32 MB
  bf16*  wqkv_bf  = (bf16*)p;  p += (size_t)3072  * 1024 * 2;  //  6 MB
  bf16*  wout_bf  = (bf16*)p;  p += (size_t)1024  * 1024 * 2;  //  2 MB
  float* qf       = (float*)p; p += (size_t)64 * 4096 * 64 * 4;// 64 MB [bh,n,d]
  float* kt       = (float*)p; p += (size_t)64 * 64 * 4096 * 4;// 64 MB [bh,d,n]
  bf16*  vt       = (bf16*)p;  p += (size_t)64 * 64 * 4096 * 2;// 32 MB [bh,e,n]
  bf16*  Et       = (bf16*)p;  p += (size_t)64 * 64 * 4096 * 2;// 32 MB [bh,d,n]
  float* Z        = (float*)p; p += (size_t)4096 * 4;          // [bh,d]
  bf16*  qb       = (bf16*)p;  p += (size_t)64 * 4096 * 64 * 2;// 32 MB [bh,n,d]
  bf16*  ctx      = (bf16*)p;  p += (size_t)64 * 64 * 64 * 2;  // [bh,d,e]
  bf16*  attn     = (bf16*)p;  p += (size_t)16384 * 1024 * 2;  // 32 MB [b,n,h*64+e]

  k_cvt<<<16384, 256, 0, stream>>>(feats, feats_bf);           // 16.7M elems
  k_cvt<<<3072,  256, 0, stream>>>(w_qkv, wqkv_bf);
  k_cvt<<<1024,  256, 0, stream>>>(w_out, wout_bf);

  k_gemm_qkv<<<dim3(128, 24), 256, 0, stream>>>(feats_bf, wqkv_bf, mask, qf, kt, vt);
  k_ksoft<<<512,   256, 0, stream>>>(kt, Et, Z);
  k_qsoft<<<32768, 256, 0, stream>>>(qf, qb);
  k_ctx<<<64,      128, 0, stream>>>(Et, vt, Z, ctx);
  k_attn<<<2048,   256, 0, stream>>>(qb, ctx, attn);
  k_gemm_out<<<dim3(128, 8), 256, 0, stream>>>(attn, wout_bf, b_out, out);
}
